// CompressiveMemory_20478404067409
// MI455X (gfx1250) — matrix operand order repaired and checked by reading, not compiled, not run on hardware
//
#include <hip/hip_runtime.h>
#include <hip/hip_bf16.h>

// ---------------------------------------------------------------------------
// CompressiveMemory on MI455X (gfx1250, wave32, WMMA bf16 path)
// B=2, S=4096, D=1024, M=4.
// All three contractions are plain bf16 GEMMs through v_wmma_f32_16x16x32_bf16
// with f32 accumulation, double-buffered LDS tiles filled by
// GLOBAL_LOAD_ASYNC_TO_LDS_B128 (no VGPR staging, ASYNCcnt-tracked).
// The retrieve softmax-weight/qnorm row scaling and the sum over the 4
// memories are folded into a single stacked-K GEMM:
//   out = A' @ Bstack,  A'[bs, m*D+d] = c[m,bs]*sigma(q),  Bstack rows m*D+d.
// ---------------------------------------------------------------------------

typedef __bf16 bf16;
typedef __attribute__((ext_vector_type(16))) __bf16 v16bf;
typedef __attribute__((ext_vector_type(8)))  float  v8f;

#define EPSV 1e-6f
#define Bdim 2
#define Sdim 4096
#define Ddim 1024
#define Mdim 4
#define Rdim 8192          // B*S
#define MDdim 4096         // M*D (stacked K axis)
#define LDS_STRIDE 40      // bf16 elems per LDS tile row (80B, bank-conflict free)

__device__ __forceinline__ float elu_p1(float x) {
    return x > 0.f ? x + 1.f : __expf(x);
}

__device__ __forceinline__ v8f wmma_bf16(v16bf a, v16bf b, v8f c) {
    return __builtin_amdgcn_wmma_f32_16x16x32_bf16(
        false, a, false, b, (short)0, c, false, false);
}

// Async-copy a 64x32 bf16 tile (row-major, gRowStride elems) into LDS
// (stride 40) -- GLOBAL_LOAD_ASYNC_TO_LDS_B128, 2 instructions per thread
// (128 threads). No VGPR data staging; tracked by ASYNCcnt.
__device__ __forceinline__ void async_tile(const bf16* g, size_t gRowStride,
                                           bf16* lds, int t) {
#pragma unroll
    for (int p = 0; p < 2; ++p) {
        int idx = t + p * 128;
        int r = idx >> 2;
        int c8 = (idx & 3) * 8;                    // elem offset, 16B chunks
        unsigned ldsAddr = (unsigned)(uintptr_t)(lds + r * LDS_STRIDE + c8);
        unsigned long long ga =
            (unsigned long long)(uintptr_t)(g + (size_t)r * gRowStride + c8);
        asm volatile("global_load_async_to_lds_b128 %0, %1, off"
                     :: "v"(ldsAddr), "v"(ga) : "memory");
    }
}

template <int N>
__device__ __forceinline__ void wait_async() {
    asm volatile("s_wait_asynccnt %0" :: "i"(N) : "memory");
}

// A-fragment: 16x32 bf16 tile. Lane l holds row (l&15); K chunk base (l>>4)*8
// in v0-3 and +16 in v4-7.
__device__ __forceinline__ v16bf load_afrag(const bf16* lds, int lane, int rowOff) {
    int r = rowOff + (lane & 15);
    int kb = (lane >> 4) * 8;
    union { uint4 u[2]; v16bf v; } f;
    f.u[0] = *(const uint4*)(lds + r * LDS_STRIDE + kb);
    f.u[1] = *(const uint4*)(lds + r * LDS_STRIDE + kb + 16);
    return f.v;
}

// B-fragment from a B^T (NxK) tile: lane l holds N=(l&15); K chunk base
// (l>>4)*8 in v0-3 and +16 in v4-7.
__device__ __forceinline__ v16bf load_bfrag(const bf16* lds, int lane, int rowOff) {
    int n = rowOff + (lane & 15);
    int kb = (lane >> 4) * 8;
    union { uint4 u[2]; v16bf v; } f;
    f.u[0] = *(const uint4*)(lds + n * LDS_STRIDE + kb);
    f.u[1] = *(const uint4*)(lds + n * LDS_STRIDE + kb + 16);
    return f.v;
}

// ---------------------------------------------------------------------------
// prep: sigma(K) -> bf16; qnorm[m][bs], knorm[bs]  (sigma(Q) recomputed later)
// ---------------------------------------------------------------------------
__global__ void prep_rows(const float* __restrict__ q, const float* __restrict__ k,
                          const float* __restrict__ norms,
                          bf16* __restrict__ sigmaK,
                          float* __restrict__ qnorm, float* __restrict__ knorm) {
    __shared__ float red[5][256];
    int bs = blockIdx.x, t = threadIdx.x;
    const float* qr = q + (size_t)bs * Ddim;
    const float* kr = k + (size_t)bs * Ddim;
    float accq[4] = {0.f, 0.f, 0.f, 0.f};
    float acck = 0.f;
#pragma unroll
    for (int i = 0; i < 4; ++i) {
        int d = t + i * 256;
        float sq = elu_p1(qr[d]);
#pragma unroll
        for (int m = 0; m < 4; ++m) accq[m] += sq * norms[m * Ddim + d];
        float sk = elu_p1(kr[d]);
        sigmaK[(size_t)bs * Ddim + d] = (bf16)sk;
        acck += sk * norms[d];   // memory_norms[0]
    }
    red[0][t] = accq[0]; red[1][t] = accq[1];
    red[2][t] = accq[2]; red[3][t] = accq[3]; red[4][t] = acck;
    __syncthreads();
    for (int s2 = 128; s2 > 0; s2 >>= 1) {
        if (t < s2)
            for (int j = 0; j < 5; ++j) red[j][t] += red[j][t + s2];
        __syncthreads();
    }
    if (t < 4) qnorm[t * Rdim + bs] = red[t][0];
    if (t == 4) knorm[bs] = red[4][0];
}

// ---------------------------------------------------------------------------
// bf16 transpose: out[c][r] = in[r][c]; grid (cols/32, rows/32), 256 threads
// ---------------------------------------------------------------------------
__global__ void transpose_bf16(const bf16* __restrict__ in, bf16* __restrict__ out,
                               int rows, int cols) {
    __shared__ bf16 tile[32][33];
    int c0 = blockIdx.x * 32, r0 = blockIdx.y * 32;
    int t = threadIdx.x;
#pragma unroll
    for (int i = 0; i < 4; ++i) {
        int idx = t + i * 256;
        int r = idx >> 5, c = idx & 31;
        tile[r][c] = in[(size_t)(r0 + r) * cols + c0 + c];
    }
    __syncthreads();
#pragma unroll
    for (int i = 0; i < 4; ++i) {
        int idx = t + i * 256;
        int r = idx >> 5, c = idx & 31;
        out[(size_t)(c0 + r) * rows + r0 + c] = tile[c][r];
    }
}

// memories (m,d,e) f32 -> memTs (e, m*D+d) bf16 stacked B^T ; grid (32,32,4)
__global__ void memTs_kernel(const float* __restrict__ mem, bf16* __restrict__ memTs) {
    __shared__ float tile[32][33];
    int e0 = blockIdx.x * 32, d0 = blockIdx.y * 32, m = blockIdx.z;
    int t = threadIdx.x;
    const float* src = mem + (size_t)m * Ddim * Ddim;
#pragma unroll
    for (int i = 0; i < 4; ++i) {
        int idx = t + i * 256;
        int d = idx >> 5, e = idx & 31;
        tile[d][e] = src[(size_t)(d0 + d) * Ddim + e0 + e];
    }
    __syncthreads();
#pragma unroll
    for (int i = 0; i < 4; ++i) {
        int idx = t + i * 256;
        int e = idx >> 5, d = idx & 31;
        memTs[(size_t)(e0 + e) * MDdim + m * Ddim + d0 + d] = (bf16)tile[d][e];
    }
}

// ---------------------------------------------------------------------------
// relevance mean + empty mask + softmax -> weights[m][b] (1 block)
// ---------------------------------------------------------------------------
__global__ void weights_kernel(const float* __restrict__ qnorm,
                               const float* __restrict__ norms,
                               float* __restrict__ weights) {
    __shared__ float red[256];
    int t = threadIdx.x;
    float absn[4], rel[4][2];
    for (int m = 0; m < 4; ++m) {
        float a = 0.f;
        for (int d = t; d < Ddim; d += 256) a += fabsf(norms[m * Ddim + d]);
        red[t] = a; __syncthreads();
        for (int s2 = 128; s2; s2 >>= 1) { if (t < s2) red[t] += red[t + s2]; __syncthreads(); }
        absn[m] = red[0]; __syncthreads();
    }
    for (int m = 0; m < 4; ++m)
        for (int b = 0; b < 2; ++b) {
            float a = 0.f;
            for (int s = t; s < Sdim; s += 256) a += qnorm[m * Rdim + b * Sdim + s];
            red[t] = a; __syncthreads();
            for (int s2 = 128; s2; s2 >>= 1) { if (t < s2) red[t] += red[t + s2]; __syncthreads(); }
            rel[m][b] = red[0] / (float)Sdim; __syncthreads();
        }
    if (t == 0) {
        for (int b = 0; b < 2; ++b) {
            float r[4], mx = -1e38f;
            for (int m = 0; m < 4; ++m) {
                r[m] = (absn[m] < EPSV) ? -1e30f : rel[m][b];
                mx = fmaxf(mx, r[m]);
            }
            float e[4], se = 0.f;
            for (int m = 0; m < 4; ++m) { e[m] = __expf(r[m] - mx); se += e[m]; }
            for (int m = 0; m < 4; ++m)
                weights[m * 2 + b] = (absn[m] < EPSV) ? 0.f : e[m] / se;
        }
    }
}

// ---------------------------------------------------------------------------
// Build scaled, m-stacked A: sigmaQs[bs][m*D+d] = c[m,bs] * elu_p1(q[bs][d])
// ---------------------------------------------------------------------------
__global__ void scaleA_kernel(const float* __restrict__ q,
                              const float* __restrict__ qnorm,
                              const float* __restrict__ weights,
                              bf16* __restrict__ sigmaQs) {
    int bs = blockIdx.x, t = threadIdx.x;
    int bidx = (bs >= Sdim) ? 1 : 0;
    float c[4];
#pragma unroll
    for (int m = 0; m < 4; ++m)
        c[m] = weights[m * 2 + bidx] / fmaxf(qnorm[m * Rdim + bs], EPSV);
    const float* qr = q + (size_t)bs * Ddim;
    bf16* dst = sigmaQs + (size_t)bs * MDdim;
#pragma unroll
    for (int i = 0; i < 4; ++i) {
        int d = t + i * 256;
        float sq = elu_p1(qr[d]);
#pragma unroll
        for (int m = 0; m < 4; ++m)
            dst[m * Ddim + d] = (bf16)(sq * c[m]);
    }
}

// ---------------------------------------------------------------------------
// retrieve: out = sigmaQs (R x 4096) @ memTs^T  -- plain stacked-K GEMM.
// 64x64 tile/block, 4 waves, 2x2 WMMA tiles/wave; async double-buffered LDS.
// ---------------------------------------------------------------------------
__global__ void retrieve_gemm(const bf16* __restrict__ sigmaQs,
                              const bf16* __restrict__ memTs,
                              float* __restrict__ out) {
    __shared__ bf16 At[2][64 * LDS_STRIDE];
    __shared__ bf16 Bt[2][64 * LDS_STRIDE];
    int t = threadIdx.x, lane = t & 31, wave = t >> 5;
    int wRow = (wave >> 1) * 32, wCol = (wave & 1) * 32;
    int rowBase = blockIdx.x * 64, colBase = blockIdx.y * 64;
    const bf16* Ag = sigmaQs + (size_t)rowBase * MDdim;
    const bf16* Bg = memTs + (size_t)colBase * MDdim;

    v8f acc[2][2];
#pragma unroll
    for (int i = 0; i < 2; ++i)
#pragma unroll
        for (int j = 0; j < 2; ++j) acc[i][j] = {};

    const int NK = MDdim / 32;
    async_tile(Ag, MDdim, At[0], t);
    async_tile(Bg, MDdim, Bt[0], t);
#pragma unroll 1
    for (int kt = 0; kt < NK; ++kt) {
        int cur = kt & 1;
        if (kt + 1 < NK) {
            async_tile(Ag + (kt + 1) * 32, MDdim, At[1 - cur], t);
            async_tile(Bg + (kt + 1) * 32, MDdim, Bt[1 - cur], t);
            wait_async<4>();
        } else {
            wait_async<0>();
        }
        __syncthreads();
        v16bf a0 = load_afrag(At[cur], lane, wRow);
        v16bf a1 = load_afrag(At[cur], lane, wRow + 16);
        v16bf b0 = load_bfrag(Bt[cur], lane, wCol);
        v16bf b1 = load_bfrag(Bt[cur], lane, wCol + 16);
        acc[0][0] = wmma_bf16(a0, b0, acc[0][0]);
        acc[0][1] = wmma_bf16(a0, b1, acc[0][1]);
        acc[1][0] = wmma_bf16(a1, b0, acc[1][0]);
        acc[1][1] = wmma_bf16(a1, b1, acc[1][1]);
        __syncthreads();
    }
#pragma unroll
    for (int i = 0; i < 2; ++i)
#pragma unroll
        for (int jj = 0; jj < 2; ++jj) {
            int col = colBase + wCol + jj * 16 + (lane & 15);
            int rbase = rowBase + wRow + i * 16 + ((lane >> 4) << 3);
#pragma unroll
            for (int j = 0; j < 8; ++j)
                out[(size_t)(rbase + j) * Ddim + col] = acc[i][jj][j];
        }
}

// ---------------------------------------------------------------------------
// update pass 1: ru = sigmaK @ mem0 ; dvT[e][bs] = values - ru/knorm (bf16)
// mem0^T tiles come from the stacked memTs (first D columns of each row).
// ---------------------------------------------------------------------------
__global__ void update1_gemm(const bf16* __restrict__ sigmaK,
                             const bf16* __restrict__ memTs,
                             const float* __restrict__ values,
                             const float* __restrict__ knorm,
                             bf16* __restrict__ dvT) {
    __shared__ bf16 At[2][64 * LDS_STRIDE];
    __shared__ bf16 Bt[2][64 * LDS_STRIDE];
    int t = threadIdx.x, lane = t & 31, wave = t >> 5;
    int wRow = (wave >> 1) * 32, wCol = (wave & 1) * 32;
    int rowBase = blockIdx.x * 64, colBase = blockIdx.y * 64;
    const bf16* Ag = sigmaK + (size_t)rowBase * Ddim;
    const bf16* Bg = memTs + (size_t)colBase * MDdim;   // m=0 segment

    v8f acc[2][2];
#pragma unroll
    for (int i = 0; i < 2; ++i)
#pragma unroll
        for (int j = 0; j < 2; ++j) acc[i][j] = {};

    const int NK = Ddim / 32;
    async_tile(Ag, Ddim, At[0], t);
    async_tile(Bg, MDdim, Bt[0], t);
#pragma unroll 1
    for (int kt = 0; kt < NK; ++kt) {
        int cur = kt & 1;
        if (kt + 1 < NK) {
            async_tile(Ag + (kt + 1) * 32, Ddim, At[1 - cur], t);
            async_tile(Bg + (kt + 1) * 32, MDdim, Bt[1 - cur], t);
            wait_async<4>();
        } else {
            wait_async<0>();
        }
        __syncthreads();
        v16bf a0 = load_afrag(At[cur], lane, wRow);
        v16bf a1 = load_afrag(At[cur], lane, wRow + 16);
        v16bf b0 = load_bfrag(Bt[cur], lane, wCol);
        v16bf b1 = load_bfrag(Bt[cur], lane, wCol + 16);
        acc[0][0] = wmma_bf16(a0, b0, acc[0][0]);
        acc[0][1] = wmma_bf16(a0, b1, acc[0][1]);
        acc[1][0] = wmma_bf16(a1, b0, acc[1][0]);
        acc[1][1] = wmma_bf16(a1, b1, acc[1][1]);
        __syncthreads();
    }
#pragma unroll
    for (int i = 0; i < 2; ++i)
#pragma unroll
        for (int jj = 0; jj < 2; ++jj) {
            int col = colBase + wCol + jj * 16 + (lane & 15);
            int rbase = rowBase + wRow + i * 16 + ((lane >> 4) << 3);
#pragma unroll
            for (int j = 0; j < 8; ++j) {
                int r = rbase + j;
                float kn = fmaxf(knorm[r], EPSV);
                float dv = values[(size_t)r * Ddim + col] - acc[i][jj][j] / kn;
                dvT[(size_t)col * Rdim + r] = (bf16)dv;
            }
        }
}

// ---------------------------------------------------------------------------
// update pass 2: new_mem[d][e] = mem0[d][e] + (sigmaK^T @ delta_v)[d][e]/(B*S)
// A = sigmaKT (D x R), B^T = dvT (D x R), K = 8192
// ---------------------------------------------------------------------------
__global__ void update2_gemm(const bf16* __restrict__ sigmaKT,
                             const bf16* __restrict__ dvT,
                             const float* __restrict__ mem0,
                             float* __restrict__ newmem) {
    __shared__ bf16 At[2][64 * LDS_STRIDE];
    __shared__ bf16 Bt[2][64 * LDS_STRIDE];
    int t = threadIdx.x, lane = t & 31, wave = t >> 5;
    int wRow = (wave >> 1) * 32, wCol = (wave & 1) * 32;
    int rowBase = blockIdx.x * 64, colBase = blockIdx.y * 64;
    const bf16* Ag = sigmaKT + (size_t)rowBase * Rdim;
    const bf16* Bg = dvT + (size_t)colBase * Rdim;

    v8f acc[2][2];
#pragma unroll
    for (int i = 0; i < 2; ++i)
#pragma unroll
        for (int j = 0; j < 2; ++j) acc[i][j] = {};

    const int NK = Rdim / 32;
    async_tile(Ag, Rdim, At[0], t);
    async_tile(Bg, Rdim, Bt[0], t);
#pragma unroll 1
    for (int kt = 0; kt < NK; ++kt) {
        int cur = kt & 1;
        if (kt + 1 < NK) {
            async_tile(Ag + (kt + 1) * 32, Rdim, At[1 - cur], t);
            async_tile(Bg + (kt + 1) * 32, Rdim, Bt[1 - cur], t);
            wait_async<4>();
        } else {
            wait_async<0>();
        }
        __syncthreads();
        v16bf a0 = load_afrag(At[cur], lane, wRow);
        v16bf a1 = load_afrag(At[cur], lane, wRow + 16);
        v16bf b0 = load_bfrag(Bt[cur], lane, wCol);
        v16bf b1 = load_bfrag(Bt[cur], lane, wCol + 16);
        acc[0][0] = wmma_bf16(a0, b0, acc[0][0]);
        acc[0][1] = wmma_bf16(a0, b1, acc[0][1]);
        acc[1][0] = wmma_bf16(a1, b0, acc[1][0]);
        acc[1][1] = wmma_bf16(a1, b1, acc[1][1]);
        __syncthreads();
    }
    const float inv = 1.0f / (float)Rdim;
#pragma unroll
    for (int i = 0; i < 2; ++i)
#pragma unroll
        for (int jj = 0; jj < 2; ++jj) {
            int col = colBase + wCol + jj * 16 + (lane & 15);
            int rbase = rowBase + wRow + i * 16 + ((lane >> 4) << 3);
#pragma unroll
            for (int j = 0; j < 8; ++j) {
                int r = rbase + j;
                newmem[(size_t)r * Ddim + col] =
                    mem0[(size_t)r * Ddim + col] + acc[i][jj][j] * inv;
            }
        }
}

// new_norm[d] = norm0[d] + sum_bs sigmaK[bs][d] / B   (via sigmaKT rows)
__global__ void norm_kernel(const bf16* __restrict__ sigmaKT,
                            const float* __restrict__ norms0,
                            float* __restrict__ outNorm) {
    __shared__ float red[256];
    int d = blockIdx.x, t = threadIdx.x;
    float a = 0.f;
    for (int r = t; r < Rdim; r += 256) a += (float)sigmaKT[(size_t)d * Rdim + r];
    red[t] = a; __syncthreads();
    for (int s2 = 128; s2; s2 >>= 1) { if (t < s2) red[t] += red[t + s2]; __syncthreads(); }
    if (t == 0) outNorm[d] = norms0[d] + red[0] * (1.0f / (float)Bdim);
}

// ---------------------------------------------------------------------------
extern "C" void kernel_launch(void* const* d_in, const int* in_sizes, int n_in,
                              void* d_out, int out_size, void* d_ws, size_t ws_size,
                              hipStream_t stream) {
    const float* q     = (const float*)d_in[0];
    const float* k     = (const float*)d_in[1];
    const float* v     = (const float*)d_in[2];
    const float* mem   = (const float*)d_in[3];
    const float* norms = (const float*)d_in[4];

    float* out     = (float*)d_out;                        // (B,S,D)
    float* newmem  = out + (size_t)Rdim * Ddim;            // (D,D)
    float* newnorm = newmem + (size_t)Ddim * Ddim;         // (D)

    char* w = (char*)d_ws;
    bf16* sigmaQs = (bf16*)w;  w += (size_t)Rdim * MDdim * 2;        // 64 MB
    bf16* sigmaK  = (bf16*)w;  w += (size_t)Rdim * Ddim * 2;         // 16 MB
    bf16* sigmaKT = (bf16*)w;  w += (size_t)Rdim * Ddim * 2;         // 16 MB
    bf16* memTs   = (bf16*)w;  w += (size_t)Ddim * MDdim * 2;        //  8 MB
    bf16* dvT     = (bf16*)w;  w += (size_t)Ddim * Rdim * 2;         // 16 MB
    float* qnorm  = (float*)w; w += (size_t)Mdim * Rdim * 4;
    float* knorm  = (float*)w; w += (size_t)Rdim * 4;
    float* wts    = (float*)w; w += 64;

    prep_rows<<<Rdim, 256, 0, stream>>>(q, k, norms, sigmaK, qnorm, knorm);
    transpose_bf16<<<dim3(Ddim / 32, Rdim / 32), 256, 0, stream>>>(sigmaK, sigmaKT, Rdim, Ddim);
    memTs_kernel<<<dim3(32, 32, Mdim), 256, 0, stream>>>(mem, memTs);
    weights_kernel<<<1, 256, 0, stream>>>(qnorm, norms, wts);
    scaleA_kernel<<<Rdim, 256, 0, stream>>>(q, qnorm, wts, sigmaQs);
    retrieve_gemm<<<dim3(Rdim / 64, Ddim / 64), 128, 0, stream>>>(sigmaQs, memTs, out);
    update1_gemm<<<dim3(Rdim / 64, Ddim / 64), 128, 0, stream>>>(sigmaK, memTs, v, knorm, dvT);
    update2_gemm<<<dim3(Ddim / 64, Ddim / 64), 128, 0, stream>>>(sigmaKT, dvT, mem, newmem);
    norm_kernel<<<Ddim, 256, 0, stream>>>(sigmaKT, norms, newnorm);
}
